// MultiHeadModel_82652350644237
// MI455X (gfx1250) — matrix operand order repaired and checked by reading, not compiled, not run on hardware
//
#include <hip/hip_runtime.h>

#define BB  4096
#define DVV 1024
#define DTT 1024
#define DD  2048   // fused dim
#define HH  1024   // hidden
#define GG  4096   // gene dim
#define EE  8      // experts

typedef __attribute__((ext_vector_type(16))) __bf16 v16bf;
typedef __attribute__((ext_vector_type(8)))  float  v8f;

union Frag { v16bf v; uint4 q[2]; };

__device__ __forceinline__ unsigned short f2bf(float f) {
    unsigned u = __float_as_uint(f);
    u += 0x7FFFu + ((u >> 16) & 1u);          // round-to-nearest-even
    return (unsigned short)(u >> 16);
}
__device__ __forceinline__ float bf2f(unsigned short h) {
    return __uint_as_float(((unsigned)h) << 16);
}

// CDNA5 async global->LDS copy (ASYNCcnt-tracked DMA, no VGPR round trip).
// ISA 15.18.3 opcode 98: copies global memory to LDS at a VGPR-held offset, 16 bytes per lane.
__device__ __forceinline__ void async_copy_b128(unsigned ldsOff, const void* gptr) {
    asm volatile("global_load_async_to_lds_b128 %0, %1, off"
                 :: "v"(ldsOff), "v"((unsigned long long)gptr)
                 : "memory");
}
__device__ __forceinline__ void wait_async0() {
    asm volatile("s_wait_asynccnt 0x0" ::: "memory");
}

// ---------------------------------------------------------------------------
// 1) Routing: histogram + prefix + scatter -> expOff[9], order[B]
// ---------------------------------------------------------------------------
__global__ __launch_bounds__(256)
void route_kernel(const int* __restrict__ organ, int* __restrict__ expOff,
                  int* __restrict__ order) {
    __shared__ int cnt[EE], off[EE + 1], cur[EE];
    const int t = threadIdx.x;
    if (t < EE) cnt[t] = 0;
    __syncthreads();
    for (int b = t; b < BB; b += 256) atomicAdd(&cnt[organ[b]], 1);
    __syncthreads();
    if (t == 0) {
        int s = 0;
        for (int e = 0; e < EE; ++e) { off[e] = s; s += cnt[e]; }
        off[EE] = s;
    }
    __syncthreads();
    if (t < EE) cur[t] = off[t];
    if (t < EE + 1) expOff[t] = off[t];
    __syncthreads();
    for (int b = t; b < BB; b += 256) {
        int e = organ[b];
        int pos = atomicAdd(&cur[e], 1);
        order[pos] = b;
    }
}

// ---------------------------------------------------------------------------
// 2) Gather + f32->bf16 convert: Xg[p][0:2048] = concat(vision,text)[order[p]]
// ---------------------------------------------------------------------------
__global__ __launch_bounds__(256)
void gather_kernel(const float* __restrict__ vision, const float* __restrict__ text,
                   const int* __restrict__ order, unsigned short* __restrict__ Xg) {
    int g  = blockIdx.x * 256 + threadIdx.x;   // one float4 (4 elems) per thread
    int p  = g >> 9;                           // 512 chunks per 2048-wide row
    int d4 = (g & 511) * 4;
    int b  = order[p];
    const float* src = (d4 < DVV) ? (vision + (size_t)b * DVV + d4)
                                  : (text   + (size_t)b * DTT + (d4 - DVV));
    float4 f = *(const float4*)src;
    unsigned h0 = f2bf(f.x), h1 = f2bf(f.y), h2 = f2bf(f.z), h3 = f2bf(f.w);
    uint2 o;
    o.x = h0 | (h1 << 16);
    o.y = h2 | (h3 << 16);
    *(uint2*)(Xg + (size_t)p * DD + d4) = o;
}

// ---------------------------------------------------------------------------
// 3) Passthrough copy of vision/text into output tail
// ---------------------------------------------------------------------------
__global__ __launch_bounds__(256)
void copy_kernel(const float4* __restrict__ v, const float4* __restrict__ t,
                 float4* __restrict__ ov, float4* __restrict__ ot) {
    int g = blockIdx.x * 256 + threadIdx.x;
    const int n = (BB * DVV) / 4;
    if (g < n) ov[g] = v[g];
    else       ot[g - n] = t[g - n];
}

// ---------------------------------------------------------------------------
// Shared tile-stage helpers for the WMMA GEMMs
//   A tile: 128 rows x 32 bf16 (already bf16 in ws)  -> async DMA to LDS
//   B tile: 32 k x 128 n f32 -> bf16, transposed [n][k], via VGPRs
// ---------------------------------------------------------------------------
__device__ __forceinline__ void stage_A_async(unsigned ldsBase,
                                              const unsigned short* __restrict__ src,
                                              int pBase, int end, int kb, int tid) {
    const int c0 = tid, c1 = tid + 256;              // 512 x 16B chunks
    int m0c = c0 >> 2, q0 = c0 & 3;
    int m1c = c1 >> 2, q1 = c1 & 3;
    int row0 = pBase + m0c; if (row0 >= end) row0 = end - 1;
    int row1 = pBase + m1c; if (row1 >= end) row1 = end - 1;
    async_copy_b128(ldsBase + c0 * 16, src + (size_t)row0 * DD + kb + q0 * 8);
    async_copy_b128(ldsBase + c1 * 16, src + (size_t)row1 * DD + kb + q1 * 8);
}

__device__ __forceinline__ void stage_B_convert(unsigned short* __restrict__ lB,
                                                const float* __restrict__ Wsel,
                                                int kb, int ldw, int colBase, int tid) {
    const int n  = tid & 127;
    const int k0 = tid >> 7;                         // 0 or 1, step 2 -> 16 k's each
    float wv[16];
#pragma unroll
    for (int u = 0; u < 16; ++u)
        wv[u] = Wsel[(size_t)(kb + k0 + u * 2) * ldw + colBase + n];
#pragma unroll
    for (int u = 0; u < 16; ++u)
        lB[n * 32 + k0 + u * 2] = f2bf(wv[u]);
}

// ---------------------------------------------------------------------------
// 4) Grouped GEMM layer 1:  H1 = relu(Xg @ W1 + b1), fused N = 2048
//    (cols [0,1024) -> immune channel, cols [1024,2048) -> gene channel)
//    block tile 128x128x32, 8 waves, each wave 64x32 = 4x2 WMMA subtiles
// ---------------------------------------------------------------------------
__global__ __launch_bounds__(256)
void gemm1_kernel(const unsigned short* __restrict__ Xg,
                  const float* __restrict__ W1_imm, const float* __restrict__ b1_imm,
                  const float* __restrict__ W1_gene, const float* __restrict__ b1_gene,
                  const int* __restrict__ expOff,
                  unsigned short* __restrict__ H1) {
    const int e     = blockIdx.z;
    const int start = expOff[e], end = expOff[e + 1];
    const int pBase = start + blockIdx.x * 128;
    if (pBase >= end) return;
    const int  nBase = blockIdx.y * 128;               // 0..2047
    const bool gene  = nBase >= HH;
    const float* Wsel = (gene ? W1_gene : W1_imm) + (size_t)e * DD * HH;
    const float* bsel = (gene ? b1_gene : b1_imm) + (size_t)e * HH;
    const int wcolBase = nBase & (HH - 1);

    __shared__ __attribute__((aligned(16))) unsigned short lA[128 * 32]; // [m][k]
    __shared__ __attribute__((aligned(16))) unsigned short lB[128 * 32]; // [n][k] transposed

    const int tid  = threadIdx.x;
    const int lane = tid & 31, wave = tid >> 5;
    const int l15  = lane & 15, kh = lane >> 4;
    const int m0w  = (wave >> 2) * 64;
    const int n0w  = (wave & 3) * 32;
    const unsigned ldsA = (unsigned)(size_t)(void*)lA;

    v8f acc[4][2] = {};

    for (int kb = 0; kb < DD; kb += 32) {
        stage_A_async(ldsA, Xg, pBase, end, kb, tid);
        stage_B_convert(lB, Wsel, kb, HH, wcolBase, tid);
        if (kb + 32 < DD)   // prefetch next K slab of weights
            __builtin_prefetch(Wsel + (size_t)(kb + 32 + (tid >> 7)) * HH + wcolBase + (tid & 127), 0, 1);
        wait_async0();
        __syncthreads();

        Frag fa[4], fb[2];
#pragma unroll
        for (int i = 0; i < 4; ++i) {
            const unsigned short* pa = lA + (m0w + i * 16 + l15) * 32 + kh * 8;
            fa[i].q[0] = *(const uint4*)pa;        // K = kh*8 .. +7
            fa[i].q[1] = *(const uint4*)(pa + 16); // K = 16+kh*8 .. +7
        }
#pragma unroll
        for (int j = 0; j < 2; ++j) {
            const unsigned short* pb = lB + (n0w + j * 16 + l15) * 32 + kh * 8;
            fb[j].q[0] = *(const uint4*)pb;        // K = kh*8 .. +7
            fb[j].q[1] = *(const uint4*)(pb + 16); // K = 16+kh*8 .. +7
        }
#pragma unroll
        for (int i = 0; i < 4; ++i)
#pragma unroll
            for (int j = 0; j < 2; ++j)
                acc[i][j] = __builtin_amdgcn_wmma_f32_16x16x32_bf16(
                    false, fa[i].v, false, fb[j].v, (short)0, acc[i][j], false, false);
        __syncthreads();
    }

    // epilogue: +bias, relu, bf16 store (grouped row order)
#pragma unroll
    for (int j = 0; j < 2; ++j) {
        const int nLoc = n0w + j * 16 + l15;
        const float bias = bsel[wcolBase + nLoc];
        const int col = nBase + nLoc;
#pragma unroll
        for (int i = 0; i < 4; ++i) {
#pragma unroll
            for (int r = 0; r < 8; ++r) {
                int m = pBase + m0w + i * 16 + kh * 8 + r;
                if (m < end) {
                    float v = acc[i][j][r] + bias;
                    H1[(size_t)m * DD + col] = f2bf(v > 0.f ? v : 0.f);
                }
            }
        }
    }
}

// ---------------------------------------------------------------------------
// 5) Grouped GEMM layer 2 (gene): out[order[p]] = H1_gene[p] @ W2_gene[e] + b2
// ---------------------------------------------------------------------------
__global__ __launch_bounds__(256)
void gemm2_kernel(const unsigned short* __restrict__ H1,
                  const float* __restrict__ W2_gene, const float* __restrict__ b2_gene,
                  const int* __restrict__ expOff, const int* __restrict__ order,
                  float* __restrict__ outGene) {
    const int e     = blockIdx.z;
    const int start = expOff[e], end = expOff[e + 1];
    const int pBase = start + blockIdx.x * 128;
    if (pBase >= end) return;
    const int nBase = blockIdx.y * 128;                // 0..4095
    const float* Wsel = W2_gene + (size_t)e * HH * GG;
    const float* bsel = b2_gene + (size_t)e * GG;

    __shared__ __attribute__((aligned(16))) unsigned short lA[128 * 32];
    __shared__ __attribute__((aligned(16))) unsigned short lB[128 * 32];
    __shared__ int lOrder[128];

    const int tid  = threadIdx.x;
    const int lane = tid & 31, wave = tid >> 5;
    const int l15  = lane & 15, kh = lane >> 4;
    const int m0w  = (wave >> 2) * 64;
    const int n0w  = (wave & 3) * 32;
    const unsigned ldsA = (unsigned)(size_t)(void*)lA;

    if (tid < 128) {
        int row = pBase + tid; if (row >= end) row = end - 1;
        lOrder[tid] = order[row];
    }

    v8f acc[4][2] = {};

    for (int kb = 0; kb < HH; kb += 32) {
        stage_A_async(ldsA, H1 + HH, pBase, end, kb, tid);   // gene cols start at 1024
        stage_B_convert(lB, Wsel, kb, GG, nBase, tid);
        if (kb + 32 < HH)
            __builtin_prefetch(Wsel + (size_t)(kb + 32 + (tid >> 7)) * GG + nBase + (tid & 127), 0, 1);
        wait_async0();
        __syncthreads();

        Frag fa[4], fb[2];
#pragma unroll
        for (int i = 0; i < 4; ++i) {
            const unsigned short* pa = lA + (m0w + i * 16 + l15) * 32 + kh * 8;
            fa[i].q[0] = *(const uint4*)pa;
            fa[i].q[1] = *(const uint4*)(pa + 16);
        }
#pragma unroll
        for (int j = 0; j < 2; ++j) {
            const unsigned short* pb = lB + (n0w + j * 16 + l15) * 32 + kh * 8;
            fb[j].q[0] = *(const uint4*)pb;
            fb[j].q[1] = *(const uint4*)(pb + 16);
        }
#pragma unroll
        for (int i = 0; i < 4; ++i)
#pragma unroll
            for (int j = 0; j < 2; ++j)
                acc[i][j] = __builtin_amdgcn_wmma_f32_16x16x32_bf16(
                    false, fa[i].v, false, fb[j].v, (short)0, acc[i][j], false, false);
        __syncthreads();
    }

    // epilogue: +bias, scatter rows back to original sample index, f32 store
#pragma unroll
    for (int j = 0; j < 2; ++j) {
        const int nLoc = n0w + j * 16 + l15;
        const int n = nBase + nLoc;
        const float bias = bsel[n];
#pragma unroll
        for (int i = 0; i < 4; ++i) {
#pragma unroll
            for (int r = 0; r < 8; ++r) {
                int m = pBase + m0w + i * 16 + kh * 8 + r;
                if (m < end) {
                    int b = lOrder[m - pBase];
                    outGene[(size_t)b * GG + n] = acc[i][j][r] + bias;
                }
            }
        }
    }
}

// ---------------------------------------------------------------------------
// 6) Immune head: one wave per sample, dot(H1_imm[p], W2_imm[e]) + b2
// ---------------------------------------------------------------------------
__global__ __launch_bounds__(256)
void immune_kernel(const unsigned short* __restrict__ H1,
                   const float* __restrict__ W2_imm, const float* __restrict__ b2_imm,
                   const int* __restrict__ expOff, const int* __restrict__ order,
                   float* __restrict__ outImm) {
    int p    = blockIdx.x * 8 + (threadIdx.x >> 5);
    int lane = threadIdx.x & 31;
    if (p >= BB) return;
    int e = 0;
#pragma unroll
    for (int i = 1; i < EE; ++i) if (p >= expOff[i]) e = i;
    const unsigned short* h = H1 + (size_t)p * DD;     // imm channel = cols [0,1024)
    const float* w = W2_imm + (size_t)e * HH;
    float s = 0.f;
    for (int k = lane; k < HH; k += 32) s += bf2f(h[k]) * w[k];
#pragma unroll
    for (int off = 16; off > 0; off >>= 1) s += __shfl_down(s, off, 32);
    if (lane == 0) outImm[order[p]] = s + b2_imm[e];
}

// ---------------------------------------------------------------------------
extern "C" void kernel_launch(void* const* d_in, const int* in_sizes, int n_in,
                              void* d_out, int out_size, void* d_ws, size_t ws_size,
                              hipStream_t stream) {
    const float* vision  = (const float*)d_in[0];
    const float* text    = (const float*)d_in[1];
    const float* W1_imm  = (const float*)d_in[2];
    const float* b1_imm  = (const float*)d_in[3];
    const float* W2_imm  = (const float*)d_in[4];
    const float* b2_imm  = (const float*)d_in[5];
    const float* W1_gene = (const float*)d_in[6];
    const float* b1_gene = (const float*)d_in[7];
    const float* W2_gene = (const float*)d_in[8];
    const float* b2_gene = (const float*)d_in[9];
    const int*   organ   = (const int*)d_in[10];

    float* out     = (float*)d_out;
    float* outImm  = out;                                // [B]
    float* outGene = out + BB;                           // [B,G]
    float* outV    = outGene + (size_t)BB * GG;          // [B,DV]
    float* outT    = outV + (size_t)BB * DVV;            // [B,DT]

    char* ws = (char*)d_ws;
    int* expOff = (int*)ws;                              // 9 ints (padded to 64B)
    int* order  = (int*)(ws + 64);                       // B ints
    unsigned short* Xg = (unsigned short*)(ws + 16640);  // 256B-aligned, B*D bf16
    unsigned short* H1 = Xg + (size_t)BB * DD;           // B*2048 bf16

    route_kernel<<<1, 256, 0, stream>>>(organ, expOff, order);
    gather_kernel<<<(BB * DD / 4) / 256, 256, 0, stream>>>(vision, text, order, Xg);
    copy_kernel<<<(2 * (BB * DVV / 4)) / 256, 256, 0, stream>>>(
        (const float4*)vision, (const float4*)text, (float4*)outV, (float4*)outT);
    gemm1_kernel<<<dim3(32, 16, 8), 256, 0, stream>>>(
        Xg, W1_imm, b1_imm, W1_gene, b1_gene, expOff, H1);
    gemm2_kernel<<<dim3(32, 32, 8), 256, 0, stream>>>(
        H1, W2_gene, b2_gene, expOff, order, outGene);
    immune_kernel<<<BB / 8, 256, 0, stream>>>(H1, W2_imm, b2_imm, expOff, order, outImm);
}
